// CausalSelfMLAttention_26182120636584
// MI455X (gfx1250) — compile-verified, matrix operand order checked by reading, not run on hardware
//
#include <hip/hip_runtime.h>
#include <hip/hip_bf16.h>

// ---------------------------------------------------------------------------
// CDNA5 (gfx1250) WMMA types
// ---------------------------------------------------------------------------
typedef __bf16 bf16_t;
typedef __attribute__((ext_vector_type(16))) __bf16 v16bf;
typedef __attribute__((ext_vector_type(8)))  __bf16 v8bf;
typedef __attribute__((ext_vector_type(8)))  float  v8f;

// Problem constants (B=8, T=2048, C=1024, H=16, dh=64, m=64, p=8, L=4)
#define TT    2048
#define CC    1024
#define HH    16
#define DH    64
#define BH    128          // B*H
#define MWIN  64
#define NB    32           // T / MWIN
#define PROWS 480          // 256+128+64+32 pooled rows per head

__device__ __forceinline__ v8f wmma_bf16(v16bf a, v16bf b, v8f c) {
  return __builtin_amdgcn_wmma_f32_16x16x32_bf16(false, a, false, b, (short)0, c,
                                                 false, false);
}

// A/B fragment for 16x16x32 bf16 WMMA from a row-major [16][>=32] bf16 array
// with leading dimension `ld`.  ISA layout (CDNA5 ISA):
//   lane l (l<16):  row = l,     K = {0..7, 16..23}
//   lane l (l>=16): row = l-16,  K = {8..15, 24..31}
// -> two contiguous 16-byte loads per lane.
__device__ __forceinline__ v16bf load_frag(const bf16_t* __restrict__ base, int ld) {
  int lane = threadIdx.x & 31;
  const bf16_t* r = base + (lane & 15) * ld + ((lane >> 4) << 3);
  v8bf lo = *(const v8bf*)(r);
  v8bf hi = *(const v8bf*)(r + 16);
  v16bf f;
#pragma unroll
  for (int i = 0; i < 8; ++i) { f[i] = lo[i]; f[i + 8] = hi[i]; }
  return f;
}

// gfx1250 async global->LDS copy (ASYNCcnt-tracked, no VGPR round trip).
// VFLAT encoding: VDST = per-lane LDS byte address, VADDR = 64-bit global addr.
__device__ __forceinline__ void async_load_b128(unsigned lds, const void* g) {
  asm volatile("global_load_async_to_lds_b128 %0, %1, off"
               :: "v"(lds), "v"(g) : "memory");
}
__device__ __forceinline__ void wait_async0() {
  asm volatile("s_wait_asynccnt 0x0" ::: "memory");
}

// ---------------------------------------------------------------------------
// Kernel 0a: f32 -> bf16 conversion of x
// ---------------------------------------------------------------------------
__global__ __launch_bounds__(256) void convert_x_kernel(const float* __restrict__ x,
                                                        bf16_t* __restrict__ xb, int n4) {
  int i = blockIdx.x * 256 + threadIdx.x;
  if (i < n4) {
    float4 v = ((const float4*)x)[i];
    bf16_t* o = xb + (size_t)i * 4;
    o[0] = (bf16_t)v.x; o[1] = (bf16_t)v.y; o[2] = (bf16_t)v.z; o[3] = (bf16_t)v.w;
  }
}

// ---------------------------------------------------------------------------
// Kernel 0b: transpose weight [K][N] f32 -> [N][K] bf16 (contiguous B frags)
// ---------------------------------------------------------------------------
__global__ __launch_bounds__(256) void transpose_w_kernel(const float* __restrict__ W,
                                                          bf16_t* __restrict__ Wt,
                                                          int K, int N) {
  size_t i = (size_t)blockIdx.x * 256 + threadIdx.x;
  if (i < (size_t)K * N) {
    int n = (int)(i / K);
    int k = (int)(i % K);
    Wt[i] = (bf16_t)W[(size_t)k * N + n];
  }
}

// ---------------------------------------------------------------------------
// Kernel 1: QKV GEMM  qkv = x @ W_attn + b_attn  -> head-major bf16 q/k/v
// ---------------------------------------------------------------------------
__global__ __launch_bounds__(256) void qkv_gemm_kernel(const bf16_t* __restrict__ xb,
                                                       const bf16_t* __restrict__ WtA,
                                                       const float* __restrict__ b_attn,
                                                       bf16_t* __restrict__ qb,
                                                       bf16_t* __restrict__ kb,
                                                       bf16_t* __restrict__ vb) {
  int wave = threadIdx.x >> 5;
  int lane = threadIdx.x & 31;
  int job  = blockIdx.x * 8 + wave;     // 1024 m-tiles * 48 n-groups = 49152
  int mtile = job / 48;
  int ngrp  = job % 48;
  int m0 = mtile * 16, n0 = ngrp * 64;
  const bf16_t* Abase = xb + (size_t)m0 * CC;
  v8f acc[4] = {};
  for (int kk = 0; kk < CC; kk += 32) {
    __builtin_prefetch(Abase + kk + 256, 0, 1);
    v16bf a = load_frag(Abase + kk, CC);
#pragma unroll
    for (int j = 0; j < 4; ++j) {
      v16bf b = load_frag(WtA + (size_t)(n0 + j * 16) * CC + kk, CC);
      acc[j] = wmma_bf16(a, b, acc[j]);
    }
  }
  int mlo = (lane >> 4) << 3;
#pragma unroll
  for (int j = 0; j < 4; ++j) {
    int n = n0 + j * 16 + (lane & 15);
    int which = n >> 10;                 // 0:q 1:k 2:v
    int c = n & (CC - 1);
    int h = c >> 6, dch = c & 63;
    bf16_t* dst = (which == 0) ? qb : (which == 1 ? kb : vb);
    float bias = b_attn[n];
#pragma unroll
    for (int r = 0; r < 8; ++r) {
      int m = m0 + mlo + r;
      int bidx = m >> 11;                // / 2048
      int t = m & (TT - 1);
      dst[((size_t)(bidx * HH + h) * TT + t) * DH + dch] = (bf16_t)(acc[j][r] + bias);
    }
  }
}

// ---------------------------------------------------------------------------
// Kernel 2: coarse avg-pool of K,V -> 4 levels (strides 8,16,32,64)
// ---------------------------------------------------------------------------
__global__ __launch_bounds__(256) void pool_kernel(const bf16_t* __restrict__ kb,
                                                   const bf16_t* __restrict__ vb,
                                                   bf16_t* __restrict__ kp,
                                                   bf16_t* __restrict__ vp) {
  int idx = blockIdx.x * 256 + threadIdx.x;
  if (idx >= BH * PROWS * DH) return;
  int dch = idx & 63;
  int row = (idx >> 6) % PROWS;
  int bh  = idx / (PROWS * DH);
  int off, s;
  if (row < 256)      { off = 0;   s = 8;  }
  else if (row < 384) { off = 256; s = 16; }
  else if (row < 448) { off = 384; s = 32; }
  else                { off = 448; s = 64; }
  int t0 = (row - off) * s;
  const bf16_t* ks = kb + ((size_t)bh * TT + t0) * DH + dch;
  const bf16_t* vs = vb + ((size_t)bh * TT + t0) * DH + dch;
  float ak = 0.f, av = 0.f;
  for (int i = 0; i < s; ++i) { ak += (float)ks[(size_t)i * DH]; av += (float)vs[(size_t)i * DH]; }
  float inv = 1.0f / (float)s;
  kp[idx] = (bf16_t)(ak * inv);
  vp[idx] = (bf16_t)(av * inv);
}

// ---------------------------------------------------------------------------
// Kernel 3: multi-level attention, one workgroup per (head, 64-query block)
// ---------------------------------------------------------------------------
__global__ __launch_bounds__(256) void attn_kernel(const bf16_t* __restrict__ qg,
                                                   const bf16_t* __restrict__ kg,
                                                   const bf16_t* __restrict__ vg,
                                                   const bf16_t* __restrict__ kp,
                                                   const bf16_t* __restrict__ vp,
                                                   bf16_t* __restrict__ O) {
  __shared__ bf16_t Qs[64 * 64];        //  8 KB   queries
  __shared__ bf16_t Ks[128 * 64];       // 16 KB   key window [prev(64)|cur(64)]
  __shared__ bf16_t Vt[64 * 128];       // 16 KB   value window TRANSPOSED [dch][j]
  __shared__ bf16_t Kc[4 * 8 * 64];     //  4 KB   coarse prev keys, 4 levels
  __shared__ bf16_t Vc[4 * 8 * 64];     //  4 KB
  __shared__ float  Sf[64 * 128];       // 32 KB   fine scores
  __shared__ float  Cc[4 * 64 * 8];     //  8 KB   coarse scores -> coarse probs
  __shared__ bf16_t Pb[64 * 128];       // 16 KB   band probabilities (bf16)

  int bh   = blockIdx.x >> 5;
  int qb_i = blockIdx.x & 31;
  int t0   = qb_i * MWIN;
  int tid  = threadIdx.x;
  int wave = tid >> 5, lane = tid & 31;

  // ---- Phase 0: global -> LDS -----------------------------------------
  {
    // Q block: 512 x 16B via async DMA to LDS
    unsigned qs_base = (unsigned)(uintptr_t)Qs;
    const bf16_t* qsrc = qg + ((size_t)bh * TT + t0) * DH;
    for (int i = tid; i < 512; i += 256)
      async_load_b128(qs_base + i * 16, qsrc + i * 8);

    // K window: prev(async or zero) + cur(async)
    unsigned ks_base = (unsigned)(uintptr_t)Ks;
    const bf16_t* kcur = kg + ((size_t)bh * TT + t0) * DH;
    uint4 z; z.x = z.y = z.z = z.w = 0u;
    for (int i = tid; i < 512; i += 256) {
      if (qb_i) async_load_b128(ks_base + i * 16, kcur - 64 * DH + i * 8);
      else      ((uint4*)Ks)[i] = z;
      async_load_b128(ks_base + (i + 512) * 16, kcur + i * 8);
    }

    // V window: load through VGPRs and transpose into Vt[dch][j]
    const bf16_t* vbase = vg + (size_t)bh * TT * DH;
    for (int c = tid; c < 1024; c += 256) {
      int j = c >> 3, dch0 = (c & 7) * 8;
      v8bf vv;
      if (j < 64 && qb_i == 0) {
#pragma unroll
        for (int e = 0; e < 8; ++e) vv[e] = (bf16_t)0.0f;
      } else {
        int t = t0 + j - 64;                     // j<64 -> prev block rows
        vv = *(const v8bf*)(vbase + (size_t)t * DH + dch0);
      }
#pragma unroll
      for (int e = 0; e < 8; ++e) Vt[(dch0 + e) * 128 + j] = vv[e];
    }

    // coarse prev keys/values: 4 levels x 8 rows x 64  (one 16B chunk/thread)
    const int off[4] = {0, 256, 384, 448};
    {
      int i = tid;                               // 256 chunks total
      int lev = i >> 6, j = i & 63;
      int cb = qb_i >> lev;                      // coarse block at level lev+1
      uint4 z2; z2.x = z2.y = z2.z = z2.w = 0u;
      uint4 uk = z2, uv = z2;
      if (cb > 0) {
        size_t base = ((size_t)bh * PROWS + off[lev] + (size_t)(cb - 1) * 8) * DH;
        uk = ((const uint4*)(kp + base))[j];
        uv = ((const uint4*)(vp + base))[j];
      }
      ((uint4*)Kc)[i] = uk;
      ((uint4*)Vc)[i] = uv;
    }
  }
  wait_async0();               // barriers do not wait ASYNCcnt
  __syncthreads();

  // ---- Phase 1: fine scores S = (Q/8) @ Kwin^T via WMMA ----------------
  {
    int j0 = wave * 16;                      // each wave owns 16 key columns
#pragma unroll
    for (int mi = 0; mi < 4; ++mi) {
      v8f acc = {};
#pragma unroll
      for (int d0 = 0; d0 < DH; d0 += 32) {
        v16bf a = load_frag(Qs + mi * 16 * DH + d0, DH);
        v16bf b = load_frag(Ks + j0 * DH + d0, DH);   // B[d][j] = K[j][d]
        acc = wmma_bf16(a, b, acc);
      }
      int mlo = (lane >> 4) << 3;
      int q = mi * 16 + mlo;
      int j = j0 + (lane & 15);
#pragma unroll
      for (int r = 0; r < 8; ++r) Sf[(q + r) * 128 + j] = acc[r] * 0.125f;
    }
  }
  __syncthreads();

  // ---- Phase 2: coarse scores (64 q x 8 keys x 4 levels) ---------------
  for (int pid = tid; pid < 512; pid += 256) {
    int q = pid >> 3, p = pid & 7;
    const bf16_t* qr = Qs + q * DH;
#pragma unroll
    for (int lev = 0; lev < 4; ++lev) {
      const bf16_t* kr = Kc + (lev * 8 + p) * DH;
      float s = 0.f;
      for (int d = 0; d < DH; ++d) s += (float)qr[d] * (float)kr[d];
      Cc[(lev * 64 + q) * 8 + p] = s * 0.125f;
    }
  }
  __syncthreads();

  // ---- Phase 3: cross-level normalization (min of per-level maxes) -----
  if (tid < 64) {
    int q = tid;
    float mn = -__builtin_inff();
    for (int j = q + 1; j <= q + 64; ++j) mn = fmaxf(mn, Sf[q * 128 + j]);
#pragma unroll
    for (int lev = 0; lev < 4; ++lev) {
      float m = -__builtin_inff();
      for (int p = 0; p < 8; ++p) m = fmaxf(m, Cc[(lev * 64 + q) * 8 + p]);
      mn = fminf(mn, m);
    }
    float esum = 0.f;
    for (int j = q + 1; j <= q + 64; ++j) esum += __expf(Sf[q * 128 + j] - mn);
    const float scl[4] = {8.f, 16.f, 32.f, 64.f};
#pragma unroll
    for (int lev = 0; lev < 4; ++lev) {
      float s = 0.f;
      for (int p = 0; p < 8; ++p) s += __expf(Cc[(lev * 64 + q) * 8 + p] - mn);
      esum += scl[lev] * s;
    }
    float inv = 1.0f / esum;
    for (int j = 0; j < 128; ++j) {
      float pr = (j > q && j <= q + 64) ? __expf(Sf[q * 128 + j] - mn) * inv : 0.f;
      Pb[q * 128 + j] = (bf16_t)pr;
    }
#pragma unroll
    for (int lev = 0; lev < 4; ++lev)
      for (int p = 0; p < 8; ++p)
        Cc[(lev * 64 + q) * 8 + p] =
            scl[lev] * __expf(Cc[(lev * 64 + q) * 8 + p] - mn) * inv;
  }
  __syncthreads();

  // ---- Phase 4: out = Pb @ Vwin (WMMA) + coarse Cc @ Vc (VALU) ---------
  {
    int b_ = bh >> 4, h = bh & 15;
#pragma unroll
    for (int i = 0; i < 2; ++i) {
      int tileid = wave * 2 + i;             // 16 tiles of 16x16
      int mi = tileid >> 2, ni = tileid & 3;
      v8f acc = {};
#pragma unroll
      for (int j0 = 0; j0 < 128; j0 += 32) {
        v16bf a = load_frag(Pb + mi * 16 * 128 + j0, 128);
        // B[k=j][n=dch] lives transposed in Vt[dch][j] -> contiguous frags
        v16bf b = load_frag(Vt + (size_t)(ni * 16) * 128 + j0, 128);
        acc = wmma_bf16(a, b, acc);
      }
      int mlo = (lane >> 4) << 3;
      int dch = ni * 16 + (lane & 15);
#pragma unroll
      for (int r = 0; r < 8; ++r) {
        int ql = mi * 16 + mlo + r;
        float v = acc[r];
#pragma unroll
        for (int lev = 0; lev < 4; ++lev)
          for (int p = 0; p < 8; ++p)
            v += Cc[(lev * 64 + ql) * 8 + p] * (float)Vc[(lev * 8 + p) * DH + dch];
        int t = t0 + ql;
        O[((size_t)(b_ * TT + t)) * CC + h * DH + dch] = (bf16_t)v;
      }
    }
  }
}

// ---------------------------------------------------------------------------
// Kernel 4: output projection  out = O @ W_proj + b_proj  (f32 result)
// ---------------------------------------------------------------------------
__global__ __launch_bounds__(256) void proj_gemm_kernel(const bf16_t* __restrict__ Ob,
                                                        const bf16_t* __restrict__ WtP,
                                                        const float* __restrict__ b_proj,
                                                        float* __restrict__ out) {
  int wave = threadIdx.x >> 5;
  int lane = threadIdx.x & 31;
  int job  = blockIdx.x * 8 + wave;       // 1024 m-tiles * 16 n-groups = 16384
  int mtile = job >> 4;
  int ngrp  = job & 15;
  int m0 = mtile * 16, n0 = ngrp * 64;
  const bf16_t* Abase = Ob + (size_t)m0 * CC;
  v8f acc[4] = {};
  for (int kk = 0; kk < CC; kk += 32) {
    __builtin_prefetch(Abase + kk + 256, 0, 1);
    v16bf a = load_frag(Abase + kk, CC);
#pragma unroll
    for (int j = 0; j < 4; ++j) {
      v16bf b = load_frag(WtP + (size_t)(n0 + j * 16) * CC + kk, CC);
      acc[j] = wmma_bf16(a, b, acc[j]);
    }
  }
  int mlo = (lane >> 4) << 3;
#pragma unroll
  for (int j = 0; j < 4; ++j) {
    int n = n0 + j * 16 + (lane & 15);
    float bias = b_proj[n];
#pragma unroll
    for (int r = 0; r < 8; ++r)
      out[(size_t)(m0 + mlo + r) * CC + n] = acc[j][r] + bias;
  }
}

// ---------------------------------------------------------------------------
// Host launcher
// ---------------------------------------------------------------------------
extern "C" void kernel_launch(void* const* d_in, const int* in_sizes, int n_in,
                              void* d_out, int out_size, void* d_ws, size_t ws_size,
                              hipStream_t stream) {
  const float* x      = (const float*)d_in[0];
  const float* W_attn = (const float*)d_in[1];
  const float* b_attn = (const float*)d_in[2];
  const float* W_proj = (const float*)d_in[3];
  const float* b_proj = (const float*)d_in[4];
  float* out = (float*)d_out;

  char* ws = (char*)d_ws;
  size_t off = 0;
  auto alloc = [&](size_t elems) -> bf16_t* {
    bf16_t* p = (bf16_t*)(ws + off);
    off = (off + elems * sizeof(bf16_t) + 255) & ~(size_t)255;
    return p;
  };
  const size_t XN = (size_t)8 * TT * CC;        // 16,777,216
  bf16_t* xb  = alloc(XN);
  bf16_t* WtA = alloc((size_t)3 * CC * CC);     // [3072][1024]
  bf16_t* WtP = alloc((size_t)CC * CC);         // [1024][1024]
  bf16_t* qb  = alloc(XN);
  bf16_t* kb  = alloc(XN);
  bf16_t* vb  = alloc(XN);
  bf16_t* kp  = alloc((size_t)BH * PROWS * DH);
  bf16_t* vp  = alloc((size_t)BH * PROWS * DH);
  bf16_t* Ob  = alloc(XN);

  // 0) precision conversion + weight transposes
  convert_x_kernel<<<(int)(XN / 4 / 256), 256, 0, stream>>>(x, xb, (int)(XN / 4));
  transpose_w_kernel<<<(int)((3u * CC * CC) / 256), 256, 0, stream>>>(W_attn, WtA, CC, 3 * CC);
  transpose_w_kernel<<<(int)(((size_t)CC * CC) / 256), 256, 0, stream>>>(W_proj, WtP, CC, CC);

  // 1) QKV projection (WMMA)
  qkv_gemm_kernel<<<6144, 256, 0, stream>>>(xb, WtA, b_attn, qb, kb, vb);

  // 2) coarse pooling
  pool_kernel<<<(BH * PROWS * DH) / 256, 256, 0, stream>>>(kb, vb, kp, vp);

  // 3) multi-level attention (WMMA QK^T and P@V, async LDS staging)
  attn_kernel<<<BH * NB, 256, 0, stream>>>(qb, kb, vb, kp, vp, Ob);

  // 4) output projection (WMMA)
  proj_gemm_kernel<<<2048, 256, 0, stream>>>(Ob, WtP, b_proj, out);
}
